// SwinTransformerBlock_39281770889719
// MI455X (gfx1250) — matrix operand order repaired and checked by reading, not compiled, not run on hardware
//
#include <hip/hip_runtime.h>
#include <hip/hip_bf16.h>

typedef __bf16 bf16_t;
typedef __attribute__((ext_vector_type(16))) __bf16 v16bf;
typedef __attribute__((ext_vector_type(8)))  float  v8f;
typedef __attribute__((ext_vector_type(4)))  unsigned int v4u;
typedef __attribute__((ext_vector_type(8)))  unsigned int v8u;
typedef __attribute__((ext_vector_type(4)))  unsigned int u32x4;
typedef __attribute__((ext_vector_type(4)))  int i32x4;
typedef __attribute__((ext_vector_type(8)))  int i32x8;

#define T_TOK   262144     // BS * 256*256
#define E_DIM   96
#define NHEADS  4
#define HDIM    24

// ---------------------------------------------------------------------------
// WMMA helpers (gfx1250 wave32, v_wmma_f32_16x16x32_bf16)
// ---------------------------------------------------------------------------
__device__ inline v8f wmma_bf16(v16bf a, v16bf b, v8f c) {
  return __builtin_amdgcn_wmma_f32_16x16x32_bf16(
      false, a, false, b, (short)0, c, false, false);
}

__device__ inline unsigned int pk2(bf16_t lo, bf16_t hi) {
  unsigned short a = __builtin_bit_cast(unsigned short, lo);
  unsigned short b = __builtin_bit_cast(unsigned short, hi);
  return (unsigned int)a | ((unsigned int)b << 16);
}

// Wave index as a *scalar* value -> wave-level guards become s_cmp/s_cbranch
// instead of exec-mask juggling with accumulator copies.
__device__ inline int wave_id() {
  return __builtin_amdgcn_readfirstlane((int)(threadIdx.x >> 5));
}

// A fragment: 16x32 bf16 tile, row-major bf16 LDS, stride ld (elems).
__device__ inline v16bf load_a_frag(const bf16_t* base, int ld, int lane) {
  int hf = lane >> 4, r = lane & 15;
  const bf16_t* p = base + r * ld + hf * 8;
  v4u lo = *(const v4u*)p;
  v4u hi = *(const v4u*)(p + 16);
  v8u u = __builtin_shufflevector(lo, hi, 0, 1, 2, 3, 4, 5, 6, 7);
  return __builtin_bit_cast(v16bf, u);
}

// B fragment: 32x16 tile staged as packed words W[kp][n]=(B[2kp][n],B[2kp+1][n]).
__device__ inline v16bf load_b_frag_pk(const unsigned int* base, int ldw, int lane) {
  int hf = lane >> 4, n = lane & 15;
  v8u u;
#pragma unroll
  for (int i = 0; i < 8; ++i) u[i] = base[((i >> 2) * 8 + hf * 4 + (i & 3)) * ldw + n];
  return __builtin_bit_cast(v16bf, u);
}

// TDM: load 64 rows x 32 bf16 tile (row stride Kelems) into LDS with 8-elem row
// padding (-> LDS stride 40 bf16), via tensor_load_to_lds (D# per ISA 8.3/8.4).
template <int KELEMS>
__device__ inline void tdm_load_a_tile(unsigned lds_off, const bf16_t* gptr) {
  unsigned long long ga = (unsigned long long)(const void*)gptr;
  u32x4 g0;
  g0[0] = 1u;                                        // count=1 (valid D#)
  g0[1] = lds_off;                                   // lds_addr (bytes)
  g0[2] = (unsigned)ga;                              // global_addr[31:0]
  g0[3] = (unsigned)((ga >> 32) & 0x01FFFFFFull) | 0x80000000u;  // addr[56:32]|type=2
  i32x8 g1;
  g1[0] = (int)0x06D10000;        // data_size=2B, pad_en, interval=16dw, amount=4dw
  g1[1] = (int)((KELEMS & 0xFFFF) << 16);            // tensor_dim0[15:0]
  g1[2] = (int)((KELEMS >> 16) & 0xFFFF);            // tensor_dim0[31:16], dim1 lo=0
  g1[3] = (int)(0x10 | (32 << 16));                  // tensor_dim1 hi (1M rows), tile_dim0=32
  g1[4] = (int)64;                                   // tile_dim1=64, tile_dim2=0
  g1[5] = (int)KELEMS;                               // tensor_dim0_stride[31:0]
  g1[6] = 0;
  g1[7] = 0;
  i32x4 z4 = {0, 0, 0, 0};
#if defined(__clang_major__) && __clang_major__ >= 23
  i32x8 z8 = {0, 0, 0, 0, 0, 0, 0, 0};
  __builtin_amdgcn_tensor_load_to_lds(g0, g1, z4, z4, z8, 0);
#else
  __builtin_amdgcn_tensor_load_to_lds(g0, g1, z4, z4, 0);
#endif
}

// ---------------------------------------------------------------------------
// Fused LayerNorm + layout permute + f32->bf16 cast. One wave per token.
// mode 0: raster -> window partition; 1: identity; 2: roll(-4,-4) shift;
// mode 3: roll(+4,+4) un-shift (+ raw f32 copy if dst_f).
// ---------------------------------------------------------------------------
__global__ __launch_bounds__(256) void ln_permute_kernel(
    const float* __restrict__ src, const float* __restrict__ gam,
    const float* __restrict__ bet, bf16_t* __restrict__ dst_bf,
    float* __restrict__ dst_f, int mode) {
  int wv = threadIdx.x >> 5, lane = threadIdx.x & 31;
  int ot = blockIdx.x * 8 + wv;
  int b = ot >> 16, r = ot & 65535;
  int w = r >> 6, p = r & 63;
  int y = ((w >> 5) << 3) + (p >> 3);
  int x = ((w & 31) << 3) + (p & 7);
  size_t s;
  if (mode == 0) {
    s = ((size_t)b << 16) + (size_t)y * 256 + x;
  } else if (mode == 1) {
    s = (size_t)ot;
  } else {
    int d = (mode == 2) ? 4 : 252;
    int yy = (y + d) & 255, xx = (x + d) & 255;
    s = ((size_t)b << 16) +
        (size_t)(((yy >> 3) << 5) + (xx >> 3)) * 64 + ((yy & 7) << 3) + (xx & 7);
  }
  const float* ps = src + s * E_DIM;
  float v0 = ps[lane], v1 = ps[lane + 32], v2 = ps[lane + 64];
  float sm = v0 + v1 + v2;
  float sq = v0 * v0 + v1 * v1 + v2 * v2;
#pragma unroll
  for (int m = 1; m < 32; m <<= 1) {
    sm += __shfl_xor(sm, m, 32);
    sq += __shfl_xor(sq, m, 32);
  }
  float mean = sm * (1.0f / 96.0f);
  float var  = sq * (1.0f / 96.0f) - mean * mean;
  float rstd = rsqrtf(var + 1e-5f);
  size_t od = (size_t)ot * E_DIM;
  if (dst_f) {
    dst_f[od + lane] = v0; dst_f[od + lane + 32] = v1; dst_f[od + lane + 64] = v2;
  }
  dst_bf[od + lane]      = (bf16_t)((v0 - mean) * rstd * gam[lane]      + bet[lane]);
  dst_bf[od + lane + 32] = (bf16_t)((v1 - mean) * rstd * gam[lane + 32] + bet[lane + 32]);
  dst_bf[od + lane + 64] = (bf16_t)((v2 - mean) * rstd * gam[lane + 64] + bet[lane + 64]);
}

// ---------------------------------------------------------------------------
// bf16 WMMA GEMM, compile-time N/K and epilogue flags.
// Block 256 (8 waves): 64 rows x full N. A tile staged by the TDM.
// ---------------------------------------------------------------------------
#define APAD 40        // A LDS row stride in bf16 (matches TDM pad fields)
#define BWPAD 4        // B LDS word-row pad: ldw % 8 == 4 -> half-wave bank split

template <int N, int K, bool RESID, bool STF, bool STB>
__global__ __launch_bounds__(256) void gemm_bf16_kernel(
    const bf16_t* __restrict__ A, const unsigned int* __restrict__ Bpk,
    const float* __restrict__ bias, const float* __restrict__ resid,
    bf16_t* __restrict__ Cb, float* __restrict__ Cf) {
  constexpr int NTILES = N / 16;
  constexpr int JT = (NTILES + 7) / 8;     // n-tiles per wave
  constexpr int LDW = N + BWPAD;
  __shared__ bf16_t       As[64 * APAD];
  __shared__ unsigned int Bsp[16 * LDW];
  const int tid = threadIdx.x, lane = tid & 31;
  const int wv = wave_id();                // scalar (uniform) wave index
  const int row0 = blockIdx.x * 64;

  v8f acc[4][JT];
#pragma unroll
  for (int i = 0; i < 4; ++i)
#pragma unroll
    for (int j = 0; j < JT; ++j) acc[i][j] = (v8f){0.f,0.f,0.f,0.f,0.f,0.f,0.f,0.f};

  const unsigned lds_a = (unsigned)(unsigned long long)(const void*)&As[0];
#pragma unroll
  for (int ks = 0; ks < K; ks += 32) {
    // stage A tile 64x32 via Tensor Data Mover (wave 0 issues, padded to APAD)
    if (wv == 0) {
      tdm_load_a_tile<K>(lds_a, A + (size_t)row0 * K + ks);
      __builtin_amdgcn_s_wait_tensorcnt(0);
    }
    // stage B tile: 16 word-rows x N words, pre-packed -> straight b128 copies
#pragma unroll
    for (int e = tid * 4; e < 16 * N; e += 256 * 4) {
      int wr = e / N, wc = e % N;
      *(uint4*)&Bsp[wr * LDW + wc] =
          *(const uint4*)&Bpk[(size_t)((ks >> 1) + wr) * N + wc];
    }
    __syncthreads();
    v16bf af[4];
#pragma unroll
    for (int mt = 0; mt < 4; ++mt) af[mt] = load_a_frag(&As[mt * 16 * APAD], APAD, lane);
#pragma unroll
    for (int j = 0; j < JT; ++j) {
      int nt = wv + j * 8;
      if (nt < NTILES) {                    // scalar guard (wv uniform)
        v16bf bfg = load_b_frag_pk(&Bsp[nt * 16], LDW, lane);
#pragma unroll
        for (int mt = 0; mt < 4; ++mt) acc[mt][j] = wmma_bf16(af[mt], bfg, acc[mt][j]);
      }
    }
    __syncthreads();
  }
  // epilogue: bias + optional residual; constexpr strides fold into offsets
  const int hf = lane >> 4, nlo = lane & 15;
#pragma unroll
  for (int j = 0; j < JT; ++j) {
    int nt = wv + j * 8;
    if (nt >= NTILES) continue;
    int n = nt * 16 + nlo;
    float bv = bias[n];
#pragma unroll
    for (int mt = 0; mt < 4; ++mt) {
      size_t base = (size_t)(row0 + mt * 16 + hf * 8) * N + n;
#pragma unroll
      for (int v = 0; v < 8; ++v) {
        float xv = acc[mt][j][v] + bv;
        if constexpr (RESID) xv += resid[base + (size_t)v * N];
        if constexpr (STF)   Cf[base + (size_t)v * N] = xv;
        if constexpr (STB)   Cb[base + (size_t)v * N] = (bf16_t)xv;
      }
    }
  }
}

// ---------------------------------------------------------------------------
// Windowed MHSA. One block per (batch,window); heads sequential.
// Shift mask computed arithmetically in registers (branch-free).
// ---------------------------------------------------------------------------
template <bool MASKED>
__global__ __launch_bounds__(256) void win_attn_kernel(
    const bf16_t* __restrict__ QKV, bf16_t* __restrict__ AO) {
  __shared__ bf16_t       Qs[64 * 40];    // A: 64x32 (dims padded)
  __shared__ unsigned int Ksp[16 * 68];   // B packed: K^T, word(dp, key)
  __shared__ unsigned int Vsp[32 * 36];   // B packed: word(kp, dim)
  __shared__ bf16_t       Ps[64 * 72];    // A: probs 64x64
  __shared__ float        Ss[64 * 68];    // f32 scores

  const int tid = threadIdx.x, lane = tid & 31;
  const int wv = wave_id();               // scalar (uniform) wave index
  const int blk = blockIdx.x;
  const size_t tok0 = (size_t)blk * 64;
  const int w = blk & 1023, wy = w >> 5, wx = w & 31;
  const unsigned short* QKVu = (const unsigned short*)QKV;
  const float scale = 0.20412414523193154f;   // 1/sqrt(24)

#pragma unroll 1
  for (int h = 0; h < NHEADS; ++h) {
    // --- stage Q (vectorized rows, pad dims 24..31) ---
    {
      int r = tid >> 2, c = tid & 3;
      uint4 qv = {0u, 0u, 0u, 0u};
      if (c < 3) qv = *(const uint4*)&QKV[(tok0 + r) * 288 + h * HDIM + c * 8];
      *(uint4*)&Qs[r * 40 + c * 8] = qv;
    }
    // --- stage K^T packed: pairs along contiguous head-dim -> single u32 loads
#pragma unroll
    for (int i = 0; i < 4; ++i) {
      int e = tid + i * 256;                 // 16 x 64 words
      int dp = e >> 6, k = e & 63;
      unsigned int wd = 0u;
      if (dp < (HDIM / 2))
        wd = *(const unsigned int*)&QKV[(tok0 + k) * 288 + 96 + h * HDIM + 2 * dp];
      Ksp[dp * 68 + k] = wd;
    }
    // --- stage V packed: pairs along key tokens ---
#pragma unroll
    for (int i = 0; i < 4; ++i) {
      int e = tid + i * 256;                 // 32 x 32 words
      int kp = e >> 5, d = e & 31;
      unsigned int wd = 0u;
      if (d < HDIM) {
        unsigned short lo = QKVu[(tok0 + 2 * kp)     * 288 + 192 + h * HDIM + d];
        unsigned short hi = QKVu[(tok0 + 2 * kp + 1) * 288 + 192 + h * HDIM + d];
        wd = (unsigned int)lo | ((unsigned int)hi << 16);
      }
      Vsp[kp * 36 + d] = wd;
    }
    __syncthreads();
    // --- scores: wave -> M-tile wv>>1, N-tiles (wv&1)*2 + {0,1} ---
    {
      int mt = wv >> 1, ntb = (wv & 1) * 2;
      v16bf aq = load_a_frag(&Qs[mt * 16 * 40], 40, lane);
      int hf = lane >> 4;
      int rowb = 0;
      if constexpr (MASKED) {
        // row group: p = mt*16+hf*8+v -> py const, px = v
        int pyr = mt * 2 + hf;
        int yyr = (wy * 8 + pyr + 4) & 255;
        rowb = ((yyr + 4) >> 3) * 32 + 1;
      }
#pragma unroll
      for (int j = 0; j < 2; ++j) {
        int nt = ntb + j;
        v16bf bk = load_b_frag_pk(&Ksp[nt * 16], 68, lane);
        v8f s = (v8f){0.f,0.f,0.f,0.f,0.f,0.f,0.f,0.f};
        s = wmma_bf16(aq, bk, s);
        int nc = nt * 16 + (lane & 15);
        int cgc = 0;
        if constexpr (MASKED) {
          int pyc = nc >> 3, pxc = nc & 7;
          int yyc = (wy * 8 + pyc + 4) & 255;
          int xxc = (wx * 8 + pxc + 4) & 255;
          cgc = ((yyc + 4) >> 3) * 32 + ((xxc + 4) >> 3) + 1;
        }
#pragma unroll
        for (int v = 0; v < 8; ++v) {
          int mr = mt * 16 + hf * 8 + v;
          float val = s[v] * scale;
          if constexpr (MASKED) {
            int xxv = (wx * 8 + v + 4) & 255;           // px = v
            int cgr = rowb + ((xxv + 4) >> 3);
            val += (cgr == cgc) ? 0.0f : -1e9f;         // branch-free cndmask
          }
          Ss[mr * 68 + nc] = val;
        }
      }
    }
    __syncthreads();
    // --- softmax: 4 threads per row, vectorized LDS traffic ---
    {
      int r = tid >> 2, q = tid & 3;
      float vbuf[16];
#pragma unroll
      for (int j = 0; j < 4; ++j) {
        float4 f = *(const float4*)&Ss[r * 68 + q * 16 + j * 4];
        vbuf[4*j] = f.x; vbuf[4*j+1] = f.y; vbuf[4*j+2] = f.z; vbuf[4*j+3] = f.w;
      }
      float mx = -1e30f;
#pragma unroll
      for (int i = 0; i < 16; ++i) mx = fmaxf(mx, vbuf[i]);
      mx = fmaxf(mx, __shfl_xor(mx, 1, 4));
      mx = fmaxf(mx, __shfl_xor(mx, 2, 4));
      float sum = 0.f;
#pragma unroll
      for (int i = 0; i < 16; ++i) { vbuf[i] = __expf(vbuf[i] - mx); sum += vbuf[i]; }
      sum += __shfl_xor(sum, 1, 4);
      sum += __shfl_xor(sum, 2, 4);
      float inv = 1.0f / sum;
      uint4 w0, w1;
      w0.x = pk2((bf16_t)(vbuf[0]*inv),  (bf16_t)(vbuf[1]*inv));
      w0.y = pk2((bf16_t)(vbuf[2]*inv),  (bf16_t)(vbuf[3]*inv));
      w0.z = pk2((bf16_t)(vbuf[4]*inv),  (bf16_t)(vbuf[5]*inv));
      w0.w = pk2((bf16_t)(vbuf[6]*inv),  (bf16_t)(vbuf[7]*inv));
      w1.x = pk2((bf16_t)(vbuf[8]*inv),  (bf16_t)(vbuf[9]*inv));
      w1.y = pk2((bf16_t)(vbuf[10]*inv), (bf16_t)(vbuf[11]*inv));
      w1.z = pk2((bf16_t)(vbuf[12]*inv), (bf16_t)(vbuf[13]*inv));
      w1.w = pk2((bf16_t)(vbuf[14]*inv), (bf16_t)(vbuf[15]*inv));
      *(uint4*)&Ps[r * 72 + q * 16]     = w0;
      *(uint4*)&Ps[r * 72 + q * 16 + 8] = w1;
    }
    __syncthreads();
    // --- O = P V : wave -> M-tile wv>>1, N-tile wv&1; K=64 in 2 steps ---
    {
      int mt = wv >> 1, nt = wv & 1;
      v8f o = (v8f){0.f,0.f,0.f,0.f,0.f,0.f,0.f,0.f};
#pragma unroll
      for (int ksx = 0; ksx < 2; ++ksx) {
        v16bf ap = load_a_frag(&Ps[mt * 16 * 72 + ksx * 32], 72, lane);
        v16bf bv = load_b_frag_pk(&Vsp[ksx * 16 * 36 + nt * 16], 36, lane);
        o = wmma_bf16(ap, bv, o);
      }
      int hf = lane >> 4, d = nt * 16 + (lane & 15);
      if (d < HDIM) {
#pragma unroll
        for (int v = 0; v < 8; ++v) {
          int p = mt * 16 + hf * 8 + v;
          AO[(tok0 + p) * 96 + h * HDIM + d] = (bf16_t)o[v];
        }
      }
    }
    __syncthreads();
  }
}

// Pack f32 weight [K x N] -> bf16 pair words: P[kp*N+n] = (W[2kp][n], W[2kp+1][n])
__global__ __launch_bounds__(256) void pack_weight_kernel(
    const float* __restrict__ W, unsigned int* __restrict__ P, int K, int N) {
  int i = blockIdx.x * 256 + threadIdx.x;
  int total = (K >> 1) * N;
  if (i < total) {
    int kp = i / N, n = i % N;
    P[i] = pk2((bf16_t)W[(size_t)(2 * kp) * N + n],
               (bf16_t)W[(size_t)(2 * kp + 1) * N + n]);
  }
}

// ---------------------------------------------------------------------------
extern "C" void kernel_launch(void* const* d_in, const int* in_sizes, int n_in,
                              void* d_out, int out_size, void* d_ws, size_t ws_size,
                              hipStream_t stream) {
  (void)in_sizes; (void)n_in; (void)out_size; (void)ws_size;
  const float* image  = (const float*)d_in[0];
  const float* g1 = (const float*)d_in[1], *b1 = (const float*)d_in[2];
  const float* g2 = (const float*)d_in[3], *b2 = (const float*)d_in[4];
  const float* g3 = (const float*)d_in[5], *b3 = (const float*)d_in[6];
  const float* g4 = (const float*)d_in[7], *b4 = (const float*)d_in[8];
  const float* qkv1_w = (const float*)d_in[9],  *qkv1_b = (const float*)d_in[10];
  const float* o1_w   = (const float*)d_in[11], *o1_b   = (const float*)d_in[12];
  const float* qkv2_w = (const float*)d_in[13], *qkv2_b = (const float*)d_in[14];
  const float* o2_w   = (const float*)d_in[15], *o2_b   = (const float*)d_in[16];
  const float* m1a_w  = (const float*)d_in[17], *m1a_b  = (const float*)d_in[18];
  const float* m1b_w  = (const float*)d_in[19], *m1b_b  = (const float*)d_in[20];
  const float* m2a_w  = (const float*)d_in[21], *m2a_b  = (const float*)d_in[22];
  const float* m2b_w  = (const float*)d_in[23], *m2b_b  = (const float*)d_in[24];
  float* out = (float*)d_out;

  size_t off = 0;
  auto carve = [&](size_t bytes) -> void* {
    void* p = (char*)d_ws + off;
    off += (bytes + 255) & ~(size_t)255;
    return p;
  };
  const size_t T = T_TOK;
  bf16_t* B1 = (bf16_t*)carve(T * 96 * 2);
  bf16_t* B2 = (bf16_t*)carve(T * 384 * 2);
  float*  F1 = (float*) carve(T * 96 * 4);
  float*  F2 = (float*) carve(T * 96 * 4);
  unsigned int* Wqkv1 = (unsigned int*)carve(48 * 288 * 4);
  unsigned int* Wo1   = (unsigned int*)carve(48 * 96 * 4);
  unsigned int* Wqkv2 = (unsigned int*)carve(48 * 288 * 4);
  unsigned int* Wo2   = (unsigned int*)carve(48 * 96 * 4);
  unsigned int* Wm1a  = (unsigned int*)carve(48 * 384 * 4);
  unsigned int* Wm1b  = (unsigned int*)carve(192 * 96 * 4);
  unsigned int* Wm2a  = (unsigned int*)carve(48 * 384 * 4);
  unsigned int* Wm2b  = (unsigned int*)carve(192 * 96 * 4);

  auto pack = [&](const float* s, unsigned int* d, int K, int N) {
    int total = (K / 2) * N;
    pack_weight_kernel<<<(total + 255) / 256, 256, 0, stream>>>(s, d, K, N);
  };
  pack(qkv1_w, Wqkv1, 96, 288); pack(o1_w, Wo1, 96, 96);
  pack(qkv2_w, Wqkv2, 96, 288); pack(o2_w, Wo2, 96, 96);
  pack(m1a_w, Wm1a, 96, 384);   pack(m1b_w, Wm1b, 384, 96);
  pack(m2a_w, Wm2a, 96, 384);   pack(m2b_w, Wm2b, 384, 96);

  const dim3 gLN(T / 8), gG(T / 64), blk(256);

  // ---- W-MSA branch ----
  ln_permute_kernel<<<gLN, blk, 0, stream>>>(image, g1, b1, B1, nullptr, 0);
  gemm_bf16_kernel<288, 96, false, false, true><<<gG, blk, 0, stream>>>(
      B1, Wqkv1, qkv1_b, nullptr, B2, nullptr);                         // QKV1
  win_attn_kernel<false><<<gG, blk, 0, stream>>>(B2, B1);               // AO1
  gemm_bf16_kernel<96, 96, true, true, false><<<gG, blk, 0, stream>>>(
      B1, Wo1, o1_b, image, nullptr, F1);                               // wm
  ln_permute_kernel<<<gLN, blk, 0, stream>>>(F1, g2, b2, B1, nullptr, 1);
  gemm_bf16_kernel<384, 96, false, false, true><<<gG, blk, 0, stream>>>(
      B1, Wm1a, m1a_b, nullptr, B2, nullptr);                           // MLP1a
  gemm_bf16_kernel<96, 384, true, true, false><<<gG, blk, 0, stream>>>(
      B2, Wm1b, m1b_b, F1, nullptr, F1);                                // o1
  // ---- SW-MSA branch ----
  ln_permute_kernel<<<gLN, blk, 0, stream>>>(F1, g3, b3, B1, nullptr, 2);
  gemm_bf16_kernel<288, 96, false, false, true><<<gG, blk, 0, stream>>>(
      B1, Wqkv2, qkv2_b, nullptr, B2, nullptr);                         // QKV2
  win_attn_kernel<true><<<gG, blk, 0, stream>>>(B2, B1);                // AO2 masked
  gemm_bf16_kernel<96, 96, false, true, false><<<gG, blk, 0, stream>>>(
      B1, Wo2, o2_b, nullptr, nullptr, F2);                             // a2
  ln_permute_kernel<<<gLN, blk, 0, stream>>>(F2, g4, b4, B1, F1, 3);    // swo + LN4
  gemm_bf16_kernel<384, 96, false, false, true><<<gG, blk, 0, stream>>>(
      B1, Wm2a, m2a_b, nullptr, B2, nullptr);                           // MLP2a
  gemm_bf16_kernel<96, 384, true, true, false><<<gG, blk, 0, stream>>>(
      B2, Wm2b, m2b_b, F1, nullptr, out);                               // out
}
